// NCA_12498354831691
// MI455X (gfx1250) — matrix operand order repaired and checked by reading, not compiled, not run on hardware
//
// CDNA5 / gfx1250 Neural-CA step.
// Pass 1: per-wave 16-pixel strip -> depthwise sobel perception (f32 stencil)
//         -> f16 WMMA GEMM 48->128 (relu) -> f16 WMMA GEMM 128->16 -> residual
//         update; writes x_new to d_out, (x_new ch3, alive0) to d_ws.
// Pass 2: 3x3 maxpool on ws ch3 -> alive gate, zeroes dead pixels in d_out.
#include <hip/hip_runtime.h>

#define NCH   16
#define HCH   128
#define BATCH 32
#define HH    256
#define WW    256
#define HW    (HH * WW)

typedef __attribute__((ext_vector_type(16))) _Float16 v16h;
typedef __attribute__((ext_vector_type(8)))  float    v8f;

struct __align__(16) U4 { unsigned int a, b, c, d; };
union Frag { U4 q[2]; v16h h; };

__global__ __launch_bounds__(256) void nca_step_kernel(
    const float* __restrict__ xin, const float* __restrict__ w1,
    const float* __restrict__ b1,  const float* __restrict__ w2,
    const unsigned char* __restrict__ umask,   // jax bool array = 1 byte/elem
    float* __restrict__ out, float* __restrict__ ws_ch3, float* __restrict__ ws_a0)
{
    // w1 staged as B-fragment layout [k(0..63)][n(0..127)], zero-padded k>=48
    __shared__ __align__(32) _Float16 s_w1b[64 * 128];   // 16 KB
    __shared__ __align__(32) _Float16 s_w2b[128 * 16];   //  4 KB
    __shared__ float                  s_b1[128];
    __shared__ __align__(32) _Float16 s_ph[8][16 * 128]; // 32 KB: p (16x64) then h (16x128)
    __shared__ float                  s_alive0[8][16];

    const int tid  = threadIdx.x;
    const int lane = tid & 31;
    const int wave = tid >> 5;
    const int m    = lane & 15;        // pixel row of the 16-pixel strip (M index)
    const int chalf = lane >> 4;       // lane half (K-offset selector)
    const int half8 = chalf * 8;

    // ---- cooperative weight staging into WMMA B-fragment layouts ----
    for (int i = tid; i < 64 * 128; i += 256) {
        int k = i >> 7, n = i & 127;
        s_w1b[i] = (k < 48) ? (_Float16)w1[n * 48 + k] : (_Float16)0.f;
    }
    for (int i = tid; i < 128 * 16; i += 256) {
        int k = i >> 4, n = i & 15;
        s_w2b[i] = (_Float16)w2[n * HCH + k];
    }
    if (tid < 128) s_b1[tid] = b1[tid];
    __syncthreads();

    // ---- strip coordinates: 16 strips/row, 4096 strips/image ----
    const int strip = blockIdx.x * 8 + wave;
    const int bimg  = strip >> 12;
    const int iy    = (strip >> 4) & 255;
    const int x0    = (strip & 15) << 4;
    const int gx    = x0 + m;

    // ---- phase 1: perception p (identity, sobel-x, sobel-y) into LDS ----
    _Float16* P = &s_ph[wave][0];                 // p layout: [m*64 + k]
    const float* xb = xin + (size_t)bimg * NCH * HW;
    #pragma unroll
    for (int t = 0; t < 8; ++t) {
        const int c = t * 2 + chalf;              // channel handled by this lane
        const float* xc = xb + (size_t)c * HW;
        float tap[3][3];
        float mx = -1e30f;
        #pragma unroll
        for (int dy = -1; dy <= 1; ++dy) {
            #pragma unroll
            for (int dx = -1; dx <= 1; ++dx) {
                int yy = iy + dy, xx = gx + dx;
                bool v = ((unsigned)yy < (unsigned)HH) && ((unsigned)xx < (unsigned)WW);
                float val = v ? xc[yy * WW + xx] : 0.f;   // conv SAME = zero pad
                tap[dy + 1][dx + 1] = val;
                if (v) mx = fmaxf(mx, val);                // maxpool ignores OOB
            }
        }
        float ident = tap[1][1];
        float sx = (-tap[0][0] + tap[0][2] - 2.f * tap[1][0] + 2.f * tap[1][2]
                    - tap[2][0] + tap[2][2]) * 0.125f;
        float sy = (-tap[0][0] - 2.f * tap[0][1] - tap[0][2]
                    + tap[2][0] + 2.f * tap[2][1] + tap[2][2]) * 0.125f;
        P[m * 64 + 3 * c + 0] = (_Float16)ident;
        P[m * 64 + 3 * c + 1] = (_Float16)sx;
        P[m * 64 + 3 * c + 2] = (_Float16)sy;
        P[m * 64 + 48 + t * 2 + chalf] = (_Float16)0.f;   // zero K-pad 48..63
        if (c == 3) s_alive0[wave][m] = (mx > 0.1f) ? 1.f : 0.f;
    }
    __syncthreads();

    // ---- load A fragments of p (16x32 f16 layout; two 16B chunks at +0/+32B) ----
    Frag a1[2];
    #pragma unroll
    for (int ks = 0; ks < 2; ++ks) {
        const U4* ap = (const U4*)&s_ph[wave][m * 64 + ks * 32 + half8];
        a1[ks].q[0] = ap[0];
        a1[ks].q[1] = ap[2];
    }

    // ---- GEMM1: h = relu(p @ w1^T + b1), 8 N-tiles x 2 K-steps ----
    _Float16* Hbuf = &s_ph[wave][0];              // h layout: [m*128 + n] (reuses p space)
    #pragma unroll
    for (int nt = 0; nt < 8; ++nt) {
        v8f acc = {};
        #pragma unroll
        for (int ks = 0; ks < 2; ++ks) {
            v16h bh;
            #pragma unroll
            for (int s = 0; s < 16; ++s)
                bh[s] = s_w1b[(ks * 32 + (s < 8 ? half8 + s : 16 + half8 + s - 8)) * 128 + nt * 16 + m];
            acc = __builtin_amdgcn_wmma_f32_16x16x32_f16(
                      false, a1[ks].h, false, bh, (short)0, acc, false, false);
        }
        const float bb = s_b1[nt * 16 + m];       // N index is lane&15 for C/D layout
        #pragma unroll
        for (int r = 0; r < 8; ++r) {
            float hv = acc[r] + bb;               // row M = r + 8*chalf
            hv = hv > 0.f ? hv : 0.f;
            Hbuf[(r + half8) * 128 + nt * 16 + m] = (_Float16)hv;
        }
    }

    // ---- GEMM2: dx = h @ w2^T, K = 128 in 4 steps ----
    v8f acc2 = {};
    #pragma unroll
    for (int ks = 0; ks < 4; ++ks) {
        Frag af;
        const U4* ap = (const U4*)&s_ph[wave][m * 128 + ks * 32 + half8];
        af.q[0] = ap[0];
        af.q[1] = ap[2];
        v16h bh;
        #pragma unroll
        for (int s = 0; s < 16; ++s)
            bh[s] = s_w2b[(ks * 32 + (s < 8 ? half8 + s : 16 + half8 + s - 8)) * 16 + m];
        acc2 = __builtin_amdgcn_wmma_f32_16x16x32_f16(
                   false, af.h, false, bh, (short)0, acc2, false, false);
    }

    // ---- epilogue: residual update; lane = out-channel, VGPR r = pixel ----
    const int ch = m;
    const size_t gbase  = ((size_t)bimg * NCH + ch) * HW + (size_t)iy * WW + x0;
    const int    pixbase = (bimg * HH + iy) * WW + x0;
    #pragma unroll
    for (int r = 0; r < 8; ++r) {
        const int mm = r + half8;
        const size_t g = gbase + mm;
        float xv = xin[g];
        float um = umask[pixbase + mm] ? 1.f : 0.f;
        float xn = xv + acc2[r] * um;
        out[g] = xn;
        if (ch == 3) {
            ws_ch3[pixbase + mm] = xn;
            ws_a0 [pixbase + mm] = s_alive0[wave][mm];
        }
    }
}

__global__ __launch_bounds__(256) void nca_alive_kernel(
    const float* __restrict__ ws_ch3, const float* __restrict__ ws_a0,
    float* __restrict__ out)
{
    const int pix = blockIdx.x * 256 + threadIdx.x;
    const int b  = pix >> 16;
    const int yy = (pix >> 8) & 255;
    const int xx = pix & 255;
    const float* c3 = ws_ch3 + ((size_t)b << 16);
    float mx = -1e30f;
    #pragma unroll
    for (int dy = -1; dy <= 1; ++dy) {
        int y2 = yy + dy;
        if ((unsigned)y2 >= (unsigned)HH) continue;
        #pragma unroll
        for (int dx = -1; dx <= 1; ++dx) {
            int x2 = xx + dx;
            if ((unsigned)x2 >= (unsigned)WW) continue;
            mx = fmaxf(mx, c3[(y2 << 8) + x2]);
        }
    }
    const bool alive = (ws_a0[pix] != 0.f) && (mx > 0.1f);
    if (!alive) {
        size_t g = ((size_t)b * NCH) * HW + (yy << 8) + xx;
        #pragma unroll
        for (int c = 0; c < NCH; ++c) out[g + (size_t)c * HW] = 0.f;
    }
}

extern "C" void kernel_launch(void* const* d_in, const int* in_sizes, int n_in,
                              void* d_out, int out_size, void* d_ws, size_t ws_size,
                              hipStream_t stream) {
    const float* x  = (const float*)d_in[0];
    // d_in[1] = sobel kernel: fixed by construction, hardcoded in the stencil.
    const float* w1 = (const float*)d_in[2];
    const float* b1 = (const float*)d_in[3];
    const float* w2 = (const float*)d_in[4];
    const unsigned char* um = (const unsigned char*)d_in[5];
    float* out = (float*)d_out;

    float* ws_ch3 = (float*)d_ws;                       // B*H*W floats (8 MB)
    float* ws_a0  = ws_ch3 + (size_t)BATCH * HW;        // B*H*W floats (8 MB)

    // 32 images * 256 rows * 16 strips = 131072 strips, 8 strips (waves) / block
    nca_step_kernel<<<131072 / 8, 256, 0, stream>>>(x, w1, b1, w2, um,
                                                    out, ws_ch3, ws_a0);
    nca_alive_kernel<<<(BATCH * HW) / 256, 256, 0, stream>>>(ws_ch3, ws_a0, out);
}
